// MHASeq2SeqCompatible_39419209843225
// MI455X (gfx1250) — matrix operand order repaired and checked by reading, not compiled, not run on hardware
//
#include <hip/hip_runtime.h>
#include <hip/hip_bf16.h>

#define B_  4
#define S_  2048
#define D_  1024
#define H_  16
#define DH_ 64

typedef __attribute__((ext_vector_type(16))) __bf16 v16bf;
typedef __attribute__((ext_vector_type(8)))  __bf16 v8bf;
typedef __attribute__((ext_vector_type(4)))  __bf16 v4bf;
typedef __attribute__((ext_vector_type(8)))  float  v8f;

// f32 -> bf16, round-to-nearest-even
static __device__ __forceinline__ __bf16 f2bf(float f) {
    unsigned u = __builtin_bit_cast(unsigned, f);
    unsigned r = (u + 0x7FFFu + ((u >> 16) & 1u)) >> 16;
    return __builtin_bit_cast(__bf16, (unsigned short)r);
}

// ---------------------------------------------------------------------------
// elementwise f32 -> bf16 (4 elements / thread)
// ---------------------------------------------------------------------------
__global__ void cvt_kernel(const float* __restrict__ in, __bf16* __restrict__ out, int n4) {
    int i = blockIdx.x * blockDim.x + threadIdx.x;
    if (i < n4) {
        float4 f = ((const float4*)in)[i];
        v4bf o;
        o[0] = f2bf(f.x); o[1] = f2bf(f.y); o[2] = f2bf(f.z); o[3] = f2bf(f.w);
        ((v4bf*)out)[i] = o;
    }
}

// ---------------------------------------------------------------------------
// WMMA fragment loaders (per CDNA5 ISA 16-bit A / B VGPR layouts, wave32)
// A (16x32, MxK): lane L holds row M=L&15; K-chunks [kb,kb+8) and [kb+16,kb+24),
//                 kb = (L>>4)*8  -> two contiguous 16B loads
// B (32x16, KxN): lane L holds column N=L&15; K-chunks [kb,kb+8) and [kb+16,kb+24), strided loads
// ---------------------------------------------------------------------------
static __device__ __forceinline__ v16bf load_a_frag(const __bf16* __restrict__ A,
                                                    int lda, int m0, int k0, int lane) {
    const __bf16* p = A + (long)(m0 + (lane & 15)) * lda + k0 + ((lane >> 4) << 3);
    v8bf lo = *(const v8bf*)p;
    v8bf hi = *(const v8bf*)(p + 16);
    v16bf r;
#pragma unroll
    for (int i = 0; i < 8; ++i) { r[i] = lo[i]; r[i + 8] = hi[i]; }
    return r;
}

static __device__ __forceinline__ v16bf load_b_frag(const __bf16* __restrict__ Bm,
                                                    int ldb, int k0, int n0, int lane) {
    const __bf16* p = Bm + (long)(k0 + ((lane >> 4) << 3)) * ldb + n0 + (lane & 15);
    v16bf r;
#pragma unroll
    for (int i = 0; i < 8; ++i) { r[i] = p[(long)i * ldb]; r[i + 8] = p[(long)(i + 16) * ldb]; }
    return r;
}

// ---------------------------------------------------------------------------
// Generic bf16 WMMA GEMM:  C[M x N] = A[M x K] * W[K x N]
// 256 threads = 8 waves; wave = 32x32 tile; block = 128x64.
// Epilogue modes:
//   0: bf16 out, [B,H,S,DH] scatter (Q / V)
//   1: bf16 out, [B,H,DH,S] scatter (K transposed for QK^T)
//   2: f32  out, row-major [M, D]   (final O projection)
// ---------------------------------------------------------------------------
__global__ __launch_bounds__(256) void gemm_wmma(const __bf16* __restrict__ A,
                                                 const __bf16* __restrict__ W,
                                                 void* __restrict__ outp,
                                                 int mode) {
    const int lane = threadIdx.x & 31;
    const int w    = threadIdx.x >> 5;
    const int m0   = blockIdx.y * 128 + (w >> 1) * 32;
    const int n0   = blockIdx.x * 64  + (w & 1)  * 32;
    const int K    = D_;

    v8f acc[2][2] = {};

    for (int k0 = 0; k0 < K; k0 += 32) {
        // prefetch next A chunk into the cache hierarchy (global_prefetch_b8)
        if (k0 + 32 < K)
            __builtin_prefetch((const void*)(A + (long)(m0 + (lane & 15)) * K + k0 + 32), 0, 1);

        v16bf a0 = load_a_frag(A, K, m0,      k0, lane);
        v16bf a1 = load_a_frag(A, K, m0 + 16, k0, lane);
        v16bf b0 = load_b_frag(W, D_, k0, n0,      lane);
        v16bf b1 = load_b_frag(W, D_, k0, n0 + 16, lane);

        acc[0][0] = __builtin_amdgcn_wmma_f32_16x16x32_bf16(false, a0, false, b0, (short)0, acc[0][0], false, false);
        acc[0][1] = __builtin_amdgcn_wmma_f32_16x16x32_bf16(false, a0, false, b1, (short)0, acc[0][1], false, false);
        acc[1][0] = __builtin_amdgcn_wmma_f32_16x16x32_bf16(false, a1, false, b0, (short)0, acc[1][0], false, false);
        acc[1][1] = __builtin_amdgcn_wmma_f32_16x16x32_bf16(false, a1, false, b1, (short)0, acc[1][1], false, false);
    }

    // C/D layout: lane holds N = n&15; VGPR v holds M = v + (lane>>4)*8
    const int hi8 = (lane >> 4) * 8;
    const int nc  = lane & 15;
#pragma unroll
    for (int i = 0; i < 2; ++i)
#pragma unroll
        for (int j = 0; j < 2; ++j)
#pragma unroll
            for (int v = 0; v < 8; ++v) {
                int m = m0 + i * 16 + v + hi8;
                int n = n0 + j * 16 + nc;
                float val = acc[i][j][v];
                if (mode == 2) {
                    ((float*)outp)[(long)m * D_ + n] = val;
                } else {
                    int b  = m >> 11, s  = m & (S_ - 1);
                    int h  = n >> 6,  dh = n & (DH_ - 1);
                    long idx = (mode == 0)
                        ? (((long)(b * H_ + h) * S_  + s ) * DH_ + dh)   // [B,H,S,DH]
                        : (((long)(b * H_ + h) * DH_ + dh) * S_  + s );  // [B,H,DH,S]
                    ((__bf16*)outp)[idx] = f2bf(val);
                }
            }
}

// ---------------------------------------------------------------------------
// Flash attention core. One wave owns a 16-query tile of one (b,h).
// Key blocks of 32 (two 16-wide WMMA score subtiles), online softmax,
// P transposed to A-fragment layout via per-wave LDS + s_wait_dscnt.
// ---------------------------------------------------------------------------
__global__ __launch_bounds__(128) void flash_attn(const __bf16* __restrict__ Q,
                                                  const __bf16* __restrict__ Kt,
                                                  const __bf16* __restrict__ V,
                                                  __bf16* __restrict__ ctx) {
    __shared__ __bf16 pbuf[4][16 * 32];

    const int lane = threadIdx.x & 31;
    const int w    = threadIdx.x >> 5;
    const int bh   = blockIdx.y;
    const int q0   = (blockIdx.x * 4 + w) * 16;
    const int b    = bh >> 4, h = bh & 15;

    const int mrow = lane & 15;
    const int hi8  = (lane >> 4) * 8;
    const int kb8  = hi8;           // A-frag K base offset

    // Q fragments for dh chunks {0,32}
    v16bf aq[2];
#pragma unroll
    for (int c = 0; c < 2; ++c) {
        const __bf16* p = Q + ((long)bh * S_ + q0 + mrow) * DH_ + c * 32 + kb8;
        v8bf lo = *(const v8bf*)p;
        v8bf hi = *(const v8bf*)(p + 16);
#pragma unroll
        for (int i = 0; i < 8; ++i) { aq[c][i] = lo[i]; aq[c][i + 8] = hi[i]; }
    }

    v8f cacc[4] = {};
    float mrun[8], lrun[8];
#pragma unroll
    for (int v = 0; v < 8; ++v) { mrun[v] = -__builtin_inff(); lrun[v] = 0.f; }

    __bf16* lds = pbuf[w];
    const int nkb = (q0 + 16 + 31) / 32;   // causal: only blocks with key0 <= q0

    for (int kb = 0; kb < nkb; ++kb) {
        const int key0 = kb * 32;

        // scores: S[16 x 32] = Q_tile @ K_tile^T, two 16-wide subtiles
        v8f sa[2] = {};
#pragma unroll
        for (int t = 0; t < 2; ++t)
#pragma unroll
            for (int c = 0; c < 2; ++c) {
                const __bf16* p = Kt + ((long)bh * DH_ + c * 32 + kb8) * S_ + key0 + t * 16 + (lane & 15);
                v16bf bk;
#pragma unroll
                for (int i = 0; i < 8; ++i) { bk[i] = p[(long)i * S_]; bk[i + 8] = p[(long)(i + 16) * S_]; }
                sa[t] = __builtin_amdgcn_wmma_f32_16x16x32_bf16(false, aq[c], false, bk, (short)0, sa[t], false, false);
            }

        // scale + causal mask + online softmax
        float p0[8], p1[8], rmax[8], rsum[8];
#pragma unroll
        for (int v = 0; v < 8; ++v) {
            int mg  = q0 + v + hi8;
            int n0g = key0 + (lane & 15);
            float s0 = sa[0][v] * 0.125f;   // 1/sqrt(DH)
            float s1 = sa[1][v] * 0.125f;
            if (n0g > mg)      s0 = -__builtin_inff();
            if (n0g + 16 > mg) s1 = -__builtin_inff();
            p0[v] = s0; p1[v] = s1;
            rmax[v] = fmaxf(s0, s1);
        }
#pragma unroll
        for (int off = 1; off < 16; off <<= 1)
#pragma unroll
            for (int v = 0; v < 8; ++v)
                rmax[v] = fmaxf(rmax[v], __shfl_xor(rmax[v], off, 32));
#pragma unroll
        for (int v = 0; v < 8; ++v) {
            float mnew  = fmaxf(mrun[v], rmax[v]);
            float alpha = __expf(mrun[v] - mnew);
            p0[v] = __expf(p0[v] - mnew);
            p1[v] = __expf(p1[v] - mnew);
            rsum[v] = p0[v] + p1[v];
            mrun[v] = mnew;
            lrun[v] *= alpha;
#pragma unroll
            for (int t = 0; t < 4; ++t) cacc[t][v] *= alpha;
        }
#pragma unroll
        for (int off = 1; off < 16; off <<= 1)
#pragma unroll
            for (int v = 0; v < 8; ++v)
                rsum[v] += __shfl_xor(rsum[v], off, 32);
#pragma unroll
        for (int v = 0; v < 8; ++v) lrun[v] += rsum[v];

        // transpose P (D-layout -> A-fragment layout) through per-wave LDS
#pragma unroll
        for (int v = 0; v < 8; ++v) {
            lds[(v + hi8) * 32 + (lane & 15)]      = f2bf(p0[v]);
            lds[(v + hi8) * 32 + 16 + (lane & 15)] = f2bf(p1[v]);
        }
        asm volatile("s_wait_dscnt 0x0" ::: "memory");   // per-wave LDS fence (CDNA5 split counter)

        v16bf ap;
        {
            const __bf16* p = lds + mrow * 32 + kb8;
            v8bf lo = *(const v8bf*)p;
            v8bf hi = *(const v8bf*)(p + 16);
#pragma unroll
            for (int i = 0; i < 8; ++i) { ap[i] = lo[i]; ap[i + 8] = hi[i]; }
        }

        // ctx += P @ V  (contraction over 32 keys)
#pragma unroll
        for (int tn = 0; tn < 4; ++tn) {
            const __bf16* p = V + ((long)bh * S_ + key0 + kb8) * DH_ + tn * 16 + (lane & 15);
            v16bf bv;
#pragma unroll
            for (int i = 0; i < 8; ++i) { bv[i] = p[i * DH_]; bv[i + 8] = p[(i + 16) * DH_]; }
            cacc[tn] = __builtin_amdgcn_wmma_f32_16x16x32_bf16(false, ap, false, bv, (short)0, cacc[tn], false, false);
        }
    }

    // normalize and store ctx as bf16, fused transpose to [B, S, H*DH]
#pragma unroll
    for (int tn = 0; tn < 4; ++tn)
#pragma unroll
        for (int v = 0; v < 8; ++v) {
            int s  = q0 + v + hi8;
            int dh = tn * 16 + (lane & 15);
            float val = cacc[tn][v] / lrun[v];
            ctx[((long)(b * S_ + s)) * D_ + h * DH_ + dh] = f2bf(val);
        }
}

// ---------------------------------------------------------------------------
// Host-side orchestration
// ---------------------------------------------------------------------------
extern "C" void kernel_launch(void* const* d_in, const int* in_sizes, int n_in,
                              void* d_out, int out_size, void* d_ws, size_t ws_size,
                              hipStream_t stream) {
    (void)in_sizes; (void)n_in; (void)out_size; (void)ws_size;

    const float* x  = (const float*)d_in[0];
    const float* Wq = (const float*)d_in[1];
    const float* Wk = (const float*)d_in[2];
    const float* Wv = (const float*)d_in[3];
    const float* Wo = (const float*)d_in[4];

    const long BSD = (long)B_ * S_ * D_;   // 8,388,608
    const long DD  = (long)D_ * D_;        // 1,048,576

    __bf16* xb  = (__bf16*)d_ws;           // bf16 shadows + intermediates (~88 MB)
    __bf16* wqb = xb  + BSD;
    __bf16* wkb = wqb + DD;
    __bf16* wvb = wkb + DD;
    __bf16* wob = wvb + DD;
    __bf16* Qb  = wob + DD;    // [B,H,S,DH]
    __bf16* Ktb = Qb  + BSD;   // [B,H,DH,S]
    __bf16* Vb  = Ktb + BSD;   // [B,H,S,DH]
    __bf16* Cb  = Vb  + BSD;   // [B,S,D]

    int n4x = (int)(BSD / 4), n4w = (int)(DD / 4);
    cvt_kernel<<<(n4x + 255) / 256, 256, 0, stream>>>(x,  xb,  n4x);
    cvt_kernel<<<(n4w + 255) / 256, 256, 0, stream>>>(Wq, wqb, n4w);
    cvt_kernel<<<(n4w + 255) / 256, 256, 0, stream>>>(Wk, wkb, n4w);
    cvt_kernel<<<(n4w + 255) / 256, 256, 0, stream>>>(Wv, wvb, n4w);
    cvt_kernel<<<(n4w + 255) / 256, 256, 0, stream>>>(Wo, wob, n4w);

    dim3 gg(D_ / 64, (B_ * S_) / 128);     // (16, 64)
    gemm_wmma<<<gg, 256, 0, stream>>>(xb, wqb, (void*)Qb,  0);
    gemm_wmma<<<gg, 256, 0, stream>>>(xb, wkb, (void*)Ktb, 1);
    gemm_wmma<<<gg, 256, 0, stream>>>(xb, wvb, (void*)Vb,  0);

    dim3 ga(S_ / 64, B_ * H_);             // (32, 64): 4 waves/block * 16 rows
    flash_attn<<<ga, 128, 0, stream>>>(Qb, Ktb, Vb, Cb);

    gemm_wmma<<<gg, 256, 0, stream>>>(Cb, wob, d_out, 2);
}
